// MCCSA_30863634989657
// MI455X (gfx1250) — compile-verified, matrix operand order checked by reading, not run on hardware
//
#include <hip/hip_runtime.h>

typedef __bf16 bf16_t;
typedef __bf16 v16bf __attribute__((ext_vector_type(16)));
typedef __bf16 v8bf  __attribute__((ext_vector_type(8)));
typedef float  v8f   __attribute__((ext_vector_type(8)));

#define BATCH 512
#define NTOK  512
#define CH    96

__device__ inline unsigned pack2bf(float a, float b) {
    union { bf16_t h[2]; unsigned u; } u;
    u.h[0] = (bf16_t)a; u.h[1] = (bf16_t)b;
    return u.u;   // lowers to v_cvt_pk_bf16_f32
}

// Load one 16-bit WMMA operand (A or B mirror layout): 8 contiguous bf16 at
// p[0..7] and 8 at p[16..23]; both halves 16B aligned -> two b128 loads.
__device__ inline v16bf load_op16(const bf16_t* p) {
    const v8bf lo = *(const v8bf*)(p);
    const v8bf hi = *(const v8bf*)(p + 16);
    v16bf r;
    #pragma unroll
    for (int j = 0; j < 8; ++j) { r[j] = lo[j]; r[8 + j] = hi[j]; }
    return r;
}

// ---------------------------------------------------------------------------
// Kernel 0: one-time f32 -> bf16 conversion of the three 96x96 pointwise
// weight matrices (27648 elements total).
// ---------------------------------------------------------------------------
__global__ __launch_bounds__(256)
void wcvt_kernel(const float* __restrict__ wq, const float* __restrict__ wk,
                 const float* __restrict__ wv, bf16_t* __restrict__ out)
{
    const int i = blockIdx.x * 256 + threadIdx.x;
    const int per = CH * CH;
    if (i < per)            out[i] = (bf16_t)wq[i];
    else if (i < 2 * per)   out[i] = (bf16_t)wk[i - per];
    else                    out[i] = (bf16_t)wv[i - 2 * per];
}

// ---------------------------------------------------------------------------
// Kernel 1: fused q/k/v pointwise projections. Each wave owns 16 rows of x:
// A-operand converted f32->bf16 ONCE, then reused for 3 tensors x 6 col-tiles
// x 3 k-chunks = 54 WMMAs. Weights read as bf16 b128 loads.
// ---------------------------------------------------------------------------
__global__ __launch_bounds__(128)
void proj_gemm_kernel(const float* __restrict__ x, const bf16_t* __restrict__ Wb,
                      const float* __restrict__ bq, const float* __restrict__ bk,
                      const float* __restrict__ bv,
                      bf16_t* __restrict__ q1, bf16_t* __restrict__ k1,
                      bf16_t* __restrict__ v1)
{
    const int lane    = threadIdx.x & 31;
    const int wave    = threadIdx.x >> 5;
    const int rowBase = blockIdx.x * 64 + wave * 16;
    const int lr      = lane & 15;
    const int khalf   = (lane >> 4) * 8;

    // A operand: 16 rows of x, 3 k-chunks of 32, converted f32 -> bf16 once
    v16bf a[3];
    const float* xr = x + (size_t)(rowBase + lr) * CH;
    #pragma unroll
    for (int c = 0; c < 3; ++c) {
        const int kb = c * 32 + khalf;
        #pragma unroll
        for (int j = 0; j < 8; ++j) {
            a[c][j]     = (bf16_t)xr[kb + j];
            a[c][8 + j] = (bf16_t)xr[kb + 16 + j];
        }
    }

    const float* biases[3] = { bq, bk, bv };
    bf16_t*      dsts[3]   = { q1, k1, v1 };

    #pragma unroll
    for (int t = 0; t < 3; ++t) {
        const bf16_t* W   = Wb + (size_t)t * CH * CH;
        const float* bias = biases[t];
        bf16_t*      dst  = dsts[t];
        #pragma unroll
        for (int ot = 0; ot < 6; ++ot) {
            const int o = ot * 16 + lr;
            const bf16_t* wr = W + (size_t)o * CH;
            v8f acc = {};
            #pragma unroll
            for (int c = 0; c < 3; ++c) {
                const v16bf bm = load_op16(wr + c * 32 + khalf);
                acc = __builtin_amdgcn_wmma_f32_16x16x32_bf16(
                          false, a[c], false, bm, (short)0, acc, false, false);
            }
            const float bb = bias[o];
            #pragma unroll
            for (int r = 0; r < 8; ++r)
                dst[(size_t)(rowBase + r + khalf) * CH + o] = (bf16_t)(acc[r] + bb);
        }
    }
}

// ---------------------------------------------------------------------------
// Kernel 2: depthwise 3x3x3 conv, SAME zero padding, groups = C.
// q/k outputs stay [B, N, C]; v output transposed to [B, C, N].
// ---------------------------------------------------------------------------
__global__ __launch_bounds__(256)
void dwconv_kernel(const bf16_t* __restrict__ q1, const bf16_t* __restrict__ k1,
                   const bf16_t* __restrict__ v1,
                   const float* __restrict__ wqd, const float* __restrict__ bqd,
                   const float* __restrict__ wkd, const float* __restrict__ bkd,
                   const float* __restrict__ wvd, const float* __restrict__ bvd,
                   bf16_t* __restrict__ qd, bf16_t* __restrict__ kd,
                   bf16_t* __restrict__ vt)
{
    const bf16_t* src; const float* W; const float* bias; bf16_t* dst; int transposed;
    if (blockIdx.y == 0)      { src = q1; W = wqd; bias = bqd; dst = qd; transposed = 0; }
    else if (blockIdx.y == 1) { src = k1; W = wkd; bias = bkd; dst = kd; transposed = 0; }
    else                      { src = v1; W = wvd; bias = bvd; dst = vt; transposed = 1; }

    const size_t idx = (size_t)blockIdx.x * 256 + threadIdx.x;
    const int c = (int)(idx % CH);
    const int n = (int)((idx / CH) % NTOK);
    const size_t b = idx / ((size_t)CH * NTOK);
    const int x0 = n & 7, y0 = (n >> 3) & 7, z0 = n >> 6;

    const float* wc = W + c * 27;
    float sum = bias[c];
    #pragma unroll
    for (int dz = -1; dz <= 1; ++dz)
        #pragma unroll
        for (int dy = -1; dy <= 1; ++dy)
            #pragma unroll
            for (int dx = -1; dx <= 1; ++dx) {
                const int z = z0 + dz, y = y0 + dy, xx = x0 + dx;
                if ((unsigned)z < 8u && (unsigned)y < 8u && (unsigned)xx < 8u) {
                    const int nn = (z << 6) | (y << 3) | xx;
                    sum += wc[(dz + 1) * 9 + (dy + 1) * 3 + (dx + 1)] *
                           (float)src[(b * NTOK + nn) * CH + c];
                }
            }
    if (transposed) dst[(b * CH + c) * NTOK + n] = (bf16_t)sum;
    else            dst[(b * NTOK + n) * CH + c] = (bf16_t)sum;
}

// ---------------------------------------------------------------------------
// Kernel 3: attention, one wave32 per (batch, 16-row tile).
// Pass 1: S = Q K^T / sqrt(C)  -> LDS f32 (prefetch next K tile)
// Pass 2: vectorized softmax; probabilities kept UNNORMALIZED in LDS bf16,
//         per-row 1/sum saved to sInv[16].
// Pass 3: O = P V, row scale applied on the f32 accumulators at store.
// ---------------------------------------------------------------------------
__global__ __launch_bounds__(32)
void attention_kernel(const bf16_t* __restrict__ qd, const bf16_t* __restrict__ kd,
                      const bf16_t* __restrict__ vt, float* __restrict__ out)
{
    __shared__ float  sS[16 * 512];   // 32 KB scores
    __shared__ bf16_t sP[16 * 512];   // 16 KB unnormalized probabilities
    __shared__ float  red[32];
    __shared__ float  sInv[16];

    const int b       = blockIdx.x >> 5;
    const int rt      = blockIdx.x & 31;
    const int rowBase = rt * 16;
    const int lane    = threadIdx.x;
    const int lr      = lane & 15;
    const int khalf   = (lane >> 4) * 8;

    const bf16_t* Q = qd + (size_t)b * NTOK * CH;  // [N, C]
    const bf16_t* K = kd + (size_t)b * NTOK * CH;  // [N, C]
    const bf16_t* V = vt + (size_t)b * CH * NTOK;  // [C, N] (transposed)

    // Q A-operand tiles
    v16bf aq[3];
    {
        const bf16_t* qr = Q + (size_t)(rowBase + lr) * CH;
        #pragma unroll
        for (int c = 0; c < 3; ++c) aq[c] = load_op16(qr + c * 32 + khalf);
    }

    const float scale = 0.1020620726f;  // 1/sqrt(96)

    // ---- Pass 1: scores ----
    for (int nt = 0; nt < 32; ++nt) {
        const bf16_t* kr = K + (size_t)(nt * 16 + lr) * CH;
        if (nt + 1 < 32)
            __builtin_prefetch(kr + 16 * CH, 0, 3);   // global_prefetch_b8
        v8f acc = {};
        #pragma unroll
        for (int c = 0; c < 3; ++c) {
            const v16bf bm = load_op16(kr + c * 32 + khalf);
            acc = __builtin_amdgcn_wmma_f32_16x16x32_bf16(
                      false, aq[c], false, bm, (short)0, acc, false, false);
        }
        #pragma unroll
        for (int r = 0; r < 8; ++r)
            sS[(r + khalf) * 512 + nt * 16 + lr] = acc[r] * scale;
    }
    __syncthreads();

    // ---- Pass 2: softmax (lane handles half of one row, float4 vectorized) ----
    {
        const int row = lr;
        const int half = lane >> 4;
        const float4* s4 = (const float4*)(sS + row * 512 + half * 256);
        uint2*        p2 = (uint2*)(sP + row * 512 + half * 256);

        float m = -3.4e38f;
        #pragma unroll 4
        for (int j = 0; j < 64; ++j) {
            const float4 v = s4[j];
            m = fmaxf(m, fmaxf(fmaxf(v.x, v.y), fmaxf(v.z, v.w)));
        }
        red[lane] = m; __syncthreads();
        m = fmaxf(red[row], red[row + 16]); __syncthreads();

        float s = 0.f;
        #pragma unroll 4
        for (int j = 0; j < 64; ++j) {
            const float4 v = s4[j];
            const float e0 = __expf(v.x - m), e1 = __expf(v.y - m);
            const float e2 = __expf(v.z - m), e3 = __expf(v.w - m);
            s += (e0 + e1) + (e2 + e3);
            uint2 pk; pk.x = pack2bf(e0, e1); pk.y = pack2bf(e2, e3);
            p2[j] = pk;
        }
        red[lane] = s; __syncthreads();
        if (lane < 16) sInv[lane] = 1.f / (red[lane] + red[lane + 16]);
    }
    __syncthreads();

    // ---- Pass 3: O = P V (P unnormalized), scale rows by sInv at the end ----
    v8f acc[6] = {};
    for (int kc = 0; kc < 16; ++kc) {
        const int kb = kc * 32 + khalf;
        const v16bf pa = load_op16(sP + lr * 512 + kb);
        #pragma unroll
        for (int ct = 0; ct < 6; ++ct) {
            const v16bf bm = load_op16(V + (size_t)(ct * 16 + lr) * NTOK + kb);
            acc[ct] = __builtin_amdgcn_wmma_f32_16x16x32_bf16(
                          false, pa, false, bm, (short)0, acc[ct], false, false);
        }
    }

    float invr[8];
    #pragma unroll
    for (int r = 0; r < 8; ++r) invr[r] = sInv[r + khalf];

    #pragma unroll
    for (int ct = 0; ct < 6; ++ct) {
        const int o = ct * 16 + lr;
        #pragma unroll
        for (int r = 0; r < 8; ++r) {
            const int row = rowBase + r + khalf;
            out[((size_t)b * NTOK + row) * CH + o] = acc[ct][r] * invr[r];
        }
    }
}

// ---------------------------------------------------------------------------
extern "C" void kernel_launch(void* const* d_in, const int* in_sizes, int n_in,
                              void* d_out, int out_size, void* d_ws, size_t ws_size,
                              hipStream_t stream)
{
    (void)in_sizes; (void)n_in; (void)out_size; (void)ws_size;

    const float* x   = (const float*)d_in[0];
    const float* wq1 = (const float*)d_in[1];
    const float* bq1 = (const float*)d_in[2];
    const float* wk1 = (const float*)d_in[3];
    const float* bk1 = (const float*)d_in[4];
    const float* wv1 = (const float*)d_in[5];
    const float* bv1 = (const float*)d_in[6];
    const float* wqd = (const float*)d_in[7];
    const float* bqd = (const float*)d_in[8];
    const float* wkd = (const float*)d_in[9];
    const float* bkd = (const float*)d_in[10];
    const float* wvd = (const float*)d_in[11];
    const float* bvd = (const float*)d_in[12];

    const size_t tensorElems = (size_t)BATCH * NTOK * CH;  // 25,165,824
    bf16_t* q1 = (bf16_t*)d_ws;
    bf16_t* k1 = q1 + tensorElems;
    bf16_t* v1 = k1 + tensorElems;
    bf16_t* qd = v1 + tensorElems;
    bf16_t* kd = qd + tensorElems;
    bf16_t* vt = kd + tensorElems;          // [B, C, N]
    bf16_t* Wb = vt + tensorElems;          // 3 x 96 x 96 bf16 weights

    // 0) one-time weight conversion (3*96*96 = 27648 elems)
    wcvt_kernel<<<dim3(108), 256, 0, stream>>>(wq1, wk1, wv1, Wb);

    // 1) fused pointwise projections (WMMA GEMM)
    proj_gemm_kernel<<<dim3((unsigned)(BATCH * NTOK / 64)), 128, 0, stream>>>(
        x, Wb, bq1, bk1, bv1, q1, k1, v1);

    // 2) depthwise 3x3x3 convs
    dim3 g2((unsigned)(tensorElems / 256), 3);
    dwconv_kernel<<<g2, 256, 0, stream>>>(q1, k1, v1,
                                          wqd, bqd, wkd, bkd, wvd, bvd,
                                          qd, kd, vt);

    // 3) attention (WMMA): one wave per (batch, 16-row tile)
    attention_kernel<<<dim3(BATCH * 32), 32, 0, stream>>>(qd, kd, vt, (float*)d_out);
}
